// TemporalRefiner_56126632624834
// MI455X (gfx1250) — compile-verified, matrix operand order checked by reading, not run on hardware
//
#include <hip/hip_runtime.h>
#include <hip/hip_bf16.h>
#include <cstdint>

typedef _Float16 v4h  __attribute__((ext_vector_type(4)));
typedef _Float16 v8h  __attribute__((ext_vector_type(8)));
typedef _Float16 v16h __attribute__((ext_vector_type(16)));
typedef float    v8f  __attribute__((ext_vector_type(8)));

#define ALPHA 0.2f
#define BETA  0.05f

__device__ __forceinline__ float siluf(float x) { return x / (1.0f + __expf(-x)); }
__device__ __forceinline__ float softplusf(float x) {
    return (x > 20.0f) ? x : log1pf(__expf(x));
}
__device__ __forceinline__ float soft_clampf(float delta, float teacher) {
    float limit = fmaxf(ALPHA * fabsf(teacher), BETA);
    return limit * tanhf(delta / (limit + 1e-6f));
}
__device__ __forceinline__ v8f v8f_zero() {
    v8f z;
#pragma unroll
    for (int i = 0; i < 8; ++i) z[i] = 0.0f;
    return z;
}
__device__ __forceinline__ v4h v4h_zero() {
    v4h z;
#pragma unroll
    for (int i = 0; i < 4; ++i) z[i] = (_Float16)0.0f;
    return z;
}

// ---------------------------------------------------------------------------
// enc conv1: 1 -> 32, 3x3 SAME, input tl [8][96][96], out f32 NCHW [8][32][9216]
// ---------------------------------------------------------------------------
__global__ void k_enc_conv1(const float* __restrict__ tl, const float* __restrict__ w,
                            const float* __restrict__ b, float* __restrict__ out) {
    int idx = blockIdx.x * blockDim.x + threadIdx.x;
    const int total = 8 * 32 * 9216;
    if (idx >= total) return;
    int p = idx % 9216;
    int oc = (idx / 9216) % 32;
    int n = idx / (9216 * 32);
    int y = p / 96, x = p % 96;
    float acc = b[oc];
#pragma unroll
    for (int ty = 0; ty < 3; ++ty) {
        int gy = y + ty - 1;
        if (gy < 0 || gy >= 96) continue;
#pragma unroll
        for (int tx = 0; tx < 3; ++tx) {
            int gx = x + tx - 1;
            if (gx < 0 || gx >= 96) continue;
            acc += w[oc * 9 + ty * 3 + tx] * tl[n * 9216 + gy * 96 + gx];
        }
    }
    out[idx] = acc;
}

// ---------------------------------------------------------------------------
// BN statistics per channel over (N, H, W).  x: NCHW.  mv[2c]=mean, mv[2c+1]=var
// ---------------------------------------------------------------------------
__global__ void k_bn_stats(const float* __restrict__ x, float* __restrict__ mv,
                           int C, int HW, int N) {
    int c = blockIdx.x;
    __shared__ float ss[256], ss2[256];
    float s = 0.0f, s2 = 0.0f;
    for (int i = threadIdx.x; i < N * HW; i += blockDim.x) {
        int n = i / HW, p = i % HW;
        float v = x[((size_t)n * C + c) * HW + p];
        s += v; s2 += v * v;
    }
    ss[threadIdx.x] = s; ss2[threadIdx.x] = s2;
    __syncthreads();
    for (int st = 128; st > 0; st >>= 1) {
        if (threadIdx.x < st) {
            ss[threadIdx.x] += ss[threadIdx.x + st];
            ss2[threadIdx.x] += ss2[threadIdx.x + st];
        }
        __syncthreads();
    }
    if (threadIdx.x == 0) {
        float inv = 1.0f / (float)(N * HW);
        float mean = ss[0] * inv;
        float var = ss2[0] * inv - mean * mean;
        mv[2 * c] = mean;
        mv[2 * c + 1] = var;
    }
}

// ---------------------------------------------------------------------------
// BN1 + ReLU, NCHW f32 -> NHWC(32) f16
// ---------------------------------------------------------------------------
__global__ void k_norm1(const float* __restrict__ x, const float* __restrict__ mv,
                        const float* __restrict__ g, const float* __restrict__ b,
                        _Float16* __restrict__ out) {
    int idx = blockIdx.x * blockDim.x + threadIdx.x;
    const int total = 8 * 32 * 9216;
    if (idx >= total) return;
    int p = idx % 9216;
    int c = (idx / 9216) % 32;
    int n = idx / (9216 * 32);
    float mean = mv[2 * c], var = mv[2 * c + 1];
    float v = g[c] * (x[idx] - mean) * rsqrtf(var + 1e-5f) + b[c];
    v = fmaxf(v, 0.0f);
    out[((size_t)n * 9216 + p) * 32 + c] = (_Float16)v;
}

// ---------------------------------------------------------------------------
// BN2 + ReLU + FiLM, NCHW f32 -> NCHW f32 (feat)
// ---------------------------------------------------------------------------
__global__ void k_norm2film(const float* __restrict__ x, const float* __restrict__ mv,
                            const float* __restrict__ g, const float* __restrict__ b,
                            const float* __restrict__ tl,
                            const float* __restrict__ fgw, const float* __restrict__ fgb,
                            const float* __restrict__ fbw, const float* __restrict__ fbb,
                            float* __restrict__ feat) {
    int idx = blockIdx.x * blockDim.x + threadIdx.x;
    const int total = 8 * 32 * 9216;
    if (idx >= total) return;
    int p = idx % 9216;
    int c = (idx / 9216) % 32;
    int n = idx / (9216 * 32);
    float mean = mv[2 * c], var = mv[2 * c + 1];
    float v = g[c] * (x[idx] - mean) * rsqrtf(var + 1e-5f) + b[c];
    v = fmaxf(v, 0.0f);
    float t = tl[n * 9216 + p];
    float gm = t * fgw[c] + fgb[c];
    float bm = t * fbw[c] + fbb[c];
    feat[idx] = v * (1.0f + gm) + bm;
}

// ---------------------------------------------------------------------------
// Weight pre-pack into WMMA B-fragment register order (f16):
// wpk[((nt*KC + kc)*32 + lane)*16 + h] = W[n][ci][ty][tx]
//   n  = nt*16 + (lane & 15)
//   kl = (h<8 ? h : h+8) + (lane>=16 ? 8 : 0)      (16-bit fragment lane map)
//   kg = kc*32 + kl ;  tap = kg / CINP ; ci = kg % CINP
// ---------------------------------------------------------------------------
__global__ void k_prepack(const float* __restrict__ w, _Float16* __restrict__ wpk,
                          int NT, int KC, int CINP, int CinReal, int CoutReal, int total) {
    int idx = blockIdx.x * blockDim.x + threadIdx.x;
    if (idx >= total) return;
    int h = idx & 15;
    int lane = (idx >> 4) & 31;
    int kc = (idx >> 9) % KC;
    int nt = idx / (512 * KC);
    int n = nt * 16 + (lane & 15);
    int kl = ((h < 8) ? h : h + 8) + ((lane >= 16) ? 8 : 0);
    int kg = kc * 32 + kl;
    int tap = kg / CINP;
    int ci = kg % CINP;
    float v = 0.0f;
    if (ci < CinReal && n < CoutReal && tap < 9)
        v = w[((size_t)(n * CinReal + ci) * 3 + tap / 3) * 3 + tap % 3];
    wpk[idx] = (_Float16)v;
}

// ---------------------------------------------------------------------------
// Implicit-GEMM 3x3 SAME conv with WMMA f16->f32.
//   in : NHWC f16, channel-padded to CINP (pad channels are zero)
//   wpk: pre-packed B fragments
//   MODE 0: out_f32 NCHW (+bias),        MODE 1: relu -> out_h16 NHWC(OutStrideC)
// grid: (W/(WAVES*16), H, N), block: WAVES*32
// ---------------------------------------------------------------------------
template <int CINP, int NT, int WAVES, int MODE>
__global__ __launch_bounds__(WAVES * 32)
void conv3x3_wmma(const _Float16* __restrict__ in, const _Float16* __restrict__ wpk,
                  const float* __restrict__ bias,
                  float* __restrict__ out_f32, _Float16* __restrict__ out_h16,
                  int H, int W, int Cout, int OutStrideC) {
    constexpr int CPT = CINP / 32;
    constexpr int KC = 9 * CPT;
    constexpr int SEG = WAVES * 16;
    constexpr int LW = SEG + 2;
    __shared__ _Float16 lds[3 * LW * CINP];

    const int tid = threadIdx.x;
    const int wave = tid >> 5;
    const int lane = tid & 31;
    const int x0 = blockIdx.x * SEG;
    const int y = blockIdx.y;
    const int img = blockIdx.z;

    // Stage halo patch (3 rows x (SEG+2) cols x CINP ch) into LDS, zero padded.
    const int nvec = (3 * LW * CINP) / 4;
    for (int i = tid; i < nvec; i += WAVES * 32) {
        int c4 = i % (CINP / 4);
        int rest = i / (CINP / 4);
        int col = rest % LW;
        int row = rest / LW;
        int gy = y - 1 + row;
        int gx = x0 - 1 + col;
        v4h v = v4h_zero();
        if (gy >= 0 && gy < H && gx >= 0 && gx < W)
            v = *(const v4h*)(in + ((size_t)(img * H + gy) * W + gx) * CINP + c4 * 4);
        *(v4h*)(lds + (row * LW + col) * CINP + c4 * 4) = v;
    }
    __syncthreads();

    const int m = lane & 15;
    const int hi = lane >> 4;  // 0 or 1
    v8f acc[NT];
#pragma unroll
    for (int nt = 0; nt < NT; ++nt) acc[nt] = v8f_zero();

#pragma unroll
    for (int kc = 0; kc < KC; ++kc) {
        const int tap = kc / CPT;
        const int ty = tap / 3;
        const int tx = tap % 3;
        const int coff = (kc % CPT) * 32;
        // A fragment: row m = output pixel, K = channel sub-chunk of this tap.
        const _Float16* ap = lds + ((size_t)(ty * LW) + (wave * 16 + m + tx)) * CINP + coff + hi * 8;
        union { v16h v; v8h h[2]; } A;
        A.h[0] = *(const v8h*)(ap);
        A.h[1] = *(const v8h*)(ap + 16);
#pragma unroll
        for (int nt = 0; nt < NT; ++nt) {
            const v16h b = *(const v16h*)(wpk + ((size_t)(nt * KC + kc) * 32 + lane) * 16);
            acc[nt] = __builtin_amdgcn_wmma_f32_16x16x32_f16(
                false, A.v, false, b, (short)0, acc[nt], false, false);
        }
    }

#pragma unroll
    for (int nt = 0; nt < NT; ++nt) {
        const int n = lane & 15;
        const int ch = nt * 16 + n;
        const float bv = bias[ch];
#pragma unroll
        for (int r = 0; r < 8; ++r) {
            const int mm = r + hi * 8;
            const int px = x0 + wave * 16 + mm;
            float v = acc[nt][r] + bv;
            if (MODE == 1) {
                v = fmaxf(v, 0.0f);
                out_h16[((size_t)(img * H + y) * W + px) * OutStrideC + ch] = (_Float16)v;
            } else {
                out_f32[(((size_t)img * Cout + ch) * H + y) * W + px] = v;
            }
        }
    }
}

// ---------------------------------------------------------------------------
// Mamba: one block (64 threads) per pixel sequence. Thread j owns d_inner ch j.
// feat / featout: f32 NCHW-style [t][32][9216]
// ---------------------------------------------------------------------------
__global__ __launch_bounds__(64)
void k_mamba(const float* __restrict__ feat,
             const float* __restrict__ in_proj_w,   // (128,32)
             const float* __restrict__ conv1d_w,    // (64,1,4)
             const float* __restrict__ conv1d_b,    // (64)
             const float* __restrict__ x_proj_w,    // (34,64)
             const float* __restrict__ dt_proj_w,   // (64,2)
             const float* __restrict__ dt_proj_b,   // (64)
             const float* __restrict__ A_log,       // (64,16)
             const float* __restrict__ Dp,          // (64)
             const float* __restrict__ out_proj_w,  // (32,64)
             float* __restrict__ featout) {
    const int pix = blockIdx.x;
    const int j = threadIdx.x;
    __shared__ float xs[32], xcs[64], xdbl[34], ys[64];

    float A[16];
#pragma unroll
    for (int s = 0; s < 16; ++s) A[s] = -__expf(A_log[j * 16 + s]);
    float h[16];
#pragma unroll
    for (int s = 0; s < 16; ++s) h[s] = 0.0f;
    float hist0 = 0.0f, hist1 = 0.0f, hist2 = 0.0f;
    const float cw0 = conv1d_w[j * 4 + 0], cw1 = conv1d_w[j * 4 + 1];
    const float cw2 = conv1d_w[j * 4 + 2], cw3 = conv1d_w[j * 4 + 3];
    const float cb = conv1d_b[j];
    const float dtb = dt_proj_b[j];
    const float dw0 = dt_proj_w[j * 2 + 0], dw1 = dt_proj_w[j * 2 + 1];
    const float dpj = Dp[j];

    for (int t = 0; t < 8; ++t) {
        __syncthreads();
        if (j < 32) xs[j] = feat[((size_t)t * 32 + j) * 9216 + pix];
        __syncthreads();

        float xi = 0.0f, z = 0.0f;
#pragma unroll
        for (int c = 0; c < 32; ++c) {
            float xv = xs[c];
            xi += in_proj_w[j * 32 + c] * xv;
            z += in_proj_w[(64 + j) * 32 + c] * xv;
        }
        // causal depthwise conv1d, kernel 4, left pad 3
        float xc = cw0 * hist0 + cw1 * hist1 + cw2 * hist2 + cw3 * xi + cb;
        hist0 = hist1; hist1 = hist2; hist2 = xi;
        xc = siluf(xc);
        xcs[j] = xc;
        __syncthreads();

        if (j < 34) {
            float a = 0.0f;
#pragma unroll
            for (int c = 0; c < 64; ++c) a += x_proj_w[j * 64 + c] * xcs[c];
            xdbl[j] = a;
        }
        __syncthreads();

        float dt = softplusf(dw0 * xdbl[0] + dw1 * xdbl[1] + dtb);
        float dtxc = dt * xc;
        float y = 0.0f;
#pragma unroll
        for (int s = 0; s < 16; ++s) {
            float dA = __expf(dt * A[s]);
            h[s] = h[s] * dA + dtxc * xdbl[2 + s];
            y += h[s] * xdbl[18 + s];
        }
        y += xc * dpj;
        y *= siluf(z);
        ys[j] = y;
        __syncthreads();

        if (j < 32) {
            float o = 0.0f;
#pragma unroll
            for (int c = 0; c < 64; ++c) o += out_proj_w[j * 64 + c] * ys[c];
            featout[((size_t)t * 32 + j) * 9216 + pix] = o;
        }
    }
}

// ---------------------------------------------------------------------------
// head conv: 32 -> 1, 3x3 SAME on 96x96.  out: delta_low slice [8][9216]
// ---------------------------------------------------------------------------
__global__ void k_head(const float* __restrict__ featout, const float* __restrict__ w,
                       const float* __restrict__ b, float* __restrict__ dlow) {
    int idx = blockIdx.x * blockDim.x + threadIdx.x;
    const int total = 8 * 9216;
    if (idx >= total) return;
    int p = idx % 9216;
    int t = idx / 9216;
    int y = p / 96, x = p % 96;
    float acc = b[0];
    for (int c = 0; c < 32; ++c) {
        const float* fp = featout + ((size_t)t * 32 + c) * 9216;
#pragma unroll
        for (int ty = 0; ty < 3; ++ty) {
            int gy = y + ty - 1;
            if (gy < 0 || gy >= 96) continue;
#pragma unroll
            for (int tx = 0; tx < 3; ++tx) {
                int gx = x + tx - 1;
                if (gx < 0 || gx >= 96) continue;
                acc += w[(c * 3 + ty) * 3 + tx] * fp[gy * 96 + gx];
            }
        }
    }
    dlow[idx] = acc;
}

// ---------------------------------------------------------------------------
// bilinear 2x upsample 96->192 (half-pixel) + soft_clamp + refined_coarse
// writes dlu slice of d_out and rc to workspace
// ---------------------------------------------------------------------------
__global__ void k_upclamp(const float* __restrict__ dlow, const float* __restrict__ tf,
                          float* __restrict__ dlu, float* __restrict__ rc) {
    int idx = blockIdx.x * blockDim.x + threadIdx.x;
    const int total = 8 * 192 * 192;
    if (idx >= total) return;
    int p = idx % (192 * 192);
    int t = idx / (192 * 192);
    int yf = p / 192, xf = p % 192;
    float sy = (yf + 0.5f) * 0.5f - 0.5f;
    float sx = (xf + 0.5f) * 0.5f - 0.5f;
    int y0 = (int)floorf(sy), x0 = (int)floorf(sx);
    float fy = sy - (float)y0, fx = sx - (float)x0;
    int y1 = min(max(y0 + 1, 0), 95), x1 = min(max(x0 + 1, 0), 95);
    y0 = min(max(y0, 0), 95); x0 = min(max(x0, 0), 95);
    const float* dl = dlow + (size_t)t * 9216;
    float v00 = dl[y0 * 96 + x0], v01 = dl[y0 * 96 + x1];
    float v10 = dl[y1 * 96 + x0], v11 = dl[y1 * 96 + x1];
    float d = (1.0f - fy) * ((1.0f - fx) * v00 + fx * v01) +
              fy * ((1.0f - fx) * v10 + fx * v11);
    dlu[idx] = d;
    float tv = tf[idx];
    rc[idx] = tv + soft_clampf(d, tv);
}

// ---------------------------------------------------------------------------
// ref1 conv: 4(->2 effective) -> 48, 3x3 SAME, ReLU, write f16 NHWC64
// (pad channels 48..63 written as zero so WMMA K-reduction is clean)
// ---------------------------------------------------------------------------
__global__ void k_ref1(const float* __restrict__ tf, const float* __restrict__ rc,
                       const float* __restrict__ w, const float* __restrict__ b,
                       _Float16* __restrict__ out) {
    int idx = blockIdx.x * blockDim.x + threadIdx.x;
    const int total = 8 * 192 * 192 * 64;
    if (idx >= total) return;
    int oc = idx & 63;
    int pix = idx >> 6;
    int p = pix % (192 * 192);
    int t = pix / (192 * 192);
    int y = p / 192, x = p % 192;
    float acc = 0.0f;
    if (oc < 48) {
        acc = b[oc];
#pragma unroll
        for (int ty = 0; ty < 3; ++ty) {
            int gy = y + ty - 1;
            if (gy < 0 || gy >= 192) continue;
#pragma unroll
            for (int tx = 0; tx < 3; ++tx) {
                int gx = x + tx - 1;
                if (gx < 0 || gx >= 192) continue;
                size_t gp = (size_t)t * 36864 + gy * 192 + gx;
                acc += w[((oc * 4 + 0) * 3 + ty) * 3 + tx] * tf[gp];
                acc += w[((oc * 4 + 1) * 3 + ty) * 3 + tx] * rc[gp];
            }
        }
        acc = fmaxf(acc, 0.0f);
    }
    out[idx] = (_Float16)acc;
}

// ---------------------------------------------------------------------------
// ref3 conv: 48 -> 1, 3x3 SAME from f16 NHWC64, + soft_clamp + final outputs
// ---------------------------------------------------------------------------
__global__ void k_ref3(const _Float16* __restrict__ r2, const float* __restrict__ w,
                       const float* __restrict__ b, const float* __restrict__ tf,
                       const float* __restrict__ rc,
                       float* __restrict__ refined, float* __restrict__ dfull) {
    int idx = blockIdx.x * blockDim.x + threadIdx.x;
    const int total = 8 * 192 * 192;
    if (idx >= total) return;
    int p = idx % (192 * 192);
    int t = idx / (192 * 192);
    int y = p / 192, x = p % 192;
    float acc = b[0];
#pragma unroll
    for (int ty = 0; ty < 3; ++ty) {
        int gy = y + ty - 1;
        if (gy < 0 || gy >= 192) continue;
#pragma unroll
        for (int tx = 0; tx < 3; ++tx) {
            int gx = x + tx - 1;
            if (gx < 0 || gx >= 192) continue;
            const _Float16* rp = r2 + ((size_t)(t * 192 + gy) * 192 + gx) * 64;
            for (int c = 0; c < 48; ++c)
                acc += w[(c * 3 + ty) * 3 + tx] * (float)rp[c];
        }
    }
    float tv = tf[idx];
    float df = soft_clampf(acc, tv);
    dfull[idx] = df;
    refined[idx] = rc[idx] + df;
}

// ---------------------------------------------------------------------------
extern "C" void kernel_launch(void* const* d_in, const int* in_sizes, int n_in,
                              void* d_out, int out_size, void* d_ws, size_t ws_size,
                              hipStream_t stream) {
    const float* tl = (const float*)d_in[0];        // (8,1,96,96)
    const float* tf = (const float*)d_in[1];        // (8,1,192,192)
    const float* enc_c1_w = (const float*)d_in[2];
    const float* enc_c1_b = (const float*)d_in[3];
    const float* enc_bn1_g = (const float*)d_in[4];
    const float* enc_bn1_b = (const float*)d_in[5];
    const float* enc_c2_w = (const float*)d_in[6];
    const float* enc_c2_b = (const float*)d_in[7];
    const float* enc_bn2_g = (const float*)d_in[8];
    const float* enc_bn2_b = (const float*)d_in[9];
    const float* film_g_w = (const float*)d_in[10];
    const float* film_g_b = (const float*)d_in[11];
    const float* film_b_w = (const float*)d_in[12];
    const float* film_b_b = (const float*)d_in[13];
    const float* in_proj_w = (const float*)d_in[14];
    const float* conv1d_w = (const float*)d_in[15];
    const float* conv1d_b = (const float*)d_in[16];
    const float* x_proj_w = (const float*)d_in[17];
    const float* dt_proj_w = (const float*)d_in[18];
    const float* dt_proj_b = (const float*)d_in[19];
    const float* A_log = (const float*)d_in[20];
    const float* Dp = (const float*)d_in[21];
    const float* out_proj_w = (const float*)d_in[22];
    const float* head_w = (const float*)d_in[23];
    const float* head_b = (const float*)d_in[24];
    const float* ref1_w = (const float*)d_in[25];
    const float* ref1_b = (const float*)d_in[26];
    const float* ref2_w = (const float*)d_in[27];
    const float* ref2_b = (const float*)d_in[28];
    const float* ref3_w = (const float*)d_in[29];
    const float* ref3_b = (const float*)d_in[30];

    float* out = (float*)d_out;
    float* out_refined = out;                       // 294912
    float* out_dlow = out + 294912;                 // 73728
    float* out_dlu = out + 368640;                  // 294912
    float* out_dfull = out + 663552;                // 294912

    // workspace carve-up (256B aligned)
    uint8_t* ws = (uint8_t*)d_ws;
    size_t off = 0;
    auto carve = [&](size_t bytes) {
        size_t o = off;
        off = (off + bytes + 255) & ~(size_t)255;
        return o;
    };
    float* f1raw = (float*)(ws + carve(8 * 32 * 9216 * 4));          // enc1 raw
    float* stats1 = (float*)(ws + carve(32 * 2 * 4));
    _Float16* e1 = (_Float16*)(ws + carve((size_t)8 * 9216 * 32 * 2));   // NHWC32 f16
    float* f2raw = (float*)(ws + carve(8 * 32 * 9216 * 4));          // enc2 raw
    float* stats2 = (float*)(ws + carve(32 * 2 * 4));
    float* feat = (float*)(ws + carve(8 * 32 * 9216 * 4));           // post BN2+FiLM
    float* featout = (float*)(ws + carve(8 * 32 * 9216 * 4));        // mamba out
    float* rc = (float*)(ws + carve((size_t)8 * 36864 * 4));         // refined_coarse
    _Float16* r1 = (_Float16*)(ws + carve((size_t)8 * 36864 * 64 * 2)); // NHWC64
    _Float16* r2 = (_Float16*)(ws + carve((size_t)8 * 36864 * 64 * 2)); // NHWC64
    _Float16* wpk2 = (_Float16*)(ws + carve(2 * 9 * 512 * 2));       // enc_c2 packed
    _Float16* wpkr2 = (_Float16*)(ws + carve(3 * 18 * 512 * 2));     // ref2 packed

    const int elems_nchw = 8 * 32 * 9216;

    // ---- encoder ----
    k_enc_conv1<<<(elems_nchw + 255) / 256, 256, 0, stream>>>(tl, enc_c1_w, enc_c1_b, f1raw);
    k_bn_stats<<<32, 256, 0, stream>>>(f1raw, stats1, 32, 9216, 8);
    k_norm1<<<(elems_nchw + 255) / 256, 256, 0, stream>>>(f1raw, stats1, enc_bn1_g, enc_bn1_b, e1);

    k_prepack<<<(2 * 9 * 512 + 255) / 256, 256, 0, stream>>>(enc_c2_w, wpk2, 2, 9, 32, 32, 32,
                                                             2 * 9 * 512);
    conv3x3_wmma<32, 2, 2, 0><<<dim3(3, 96, 8), 64, 0, stream>>>(
        e1, wpk2, enc_c2_b, f2raw, nullptr, 96, 96, 32, 0);

    k_bn_stats<<<32, 256, 0, stream>>>(f2raw, stats2, 32, 9216, 8);
    k_norm2film<<<(elems_nchw + 255) / 256, 256, 0, stream>>>(
        f2raw, stats2, enc_bn2_g, enc_bn2_b, tl, film_g_w, film_g_b, film_b_w, film_b_b, feat);

    // ---- mamba over 9216 pixel sequences ----
    k_mamba<<<9216, 64, 0, stream>>>(feat, in_proj_w, conv1d_w, conv1d_b, x_proj_w,
                                     dt_proj_w, dt_proj_b, A_log, Dp, out_proj_w, featout);

    // ---- head + upsample + clamp ----
    k_head<<<(8 * 9216 + 255) / 256, 256, 0, stream>>>(featout, head_w, head_b, out_dlow);
    k_upclamp<<<(8 * 36864 + 255) / 256, 256, 0, stream>>>(out_dlow, tf, out_dlu, rc);

    // ---- refiner ----
    k_ref1<<<((size_t)8 * 36864 * 64 + 255) / 256, 256, 0, stream>>>(tf, rc, ref1_w, ref1_b, r1);
    k_prepack<<<(3 * 18 * 512 + 255) / 256, 256, 0, stream>>>(ref2_w, wpkr2, 3, 18, 64, 48, 48,
                                                              3 * 18 * 512);
    conv3x3_wmma<64, 3, 4, 1><<<dim3(3, 192, 8), 128, 0, stream>>>(
        r1, wpkr2, ref2_b, nullptr, r2, 192, 192, 48, 64);
    k_ref3<<<(8 * 36864 + 255) / 256, 256, 0, stream>>>(r2, ref3_w, ref3_b, tf, rc,
                                                        out_refined, out_dfull);
}
